// TriangleMultiplicativeModule_6493990551929
// MI455X (gfx1250) — compile-verified, matrix operand order checked by reading, not run on hardware
//
#include <hip/hip_runtime.h>
#include <hip/hip_bf16.h>

typedef __attribute__((ext_vector_type(16))) __bf16 v16bf;
typedef __attribute__((ext_vector_type(8)))  float  v8f;

#define NSEQ 512
#define DIM  128
#define NN   (NSEQ * NSEQ)   // 262144 rows

struct __align__(8) US4 { unsigned short x, y, z, w; };

__device__ __forceinline__ unsigned short f2bf(float f) {
    unsigned u = __builtin_bit_cast(unsigned, f);
    u += 0x7FFFu + ((u >> 16) & 1u);       // round-to-nearest-even
    return (unsigned short)(u >> 16);
}
__device__ __forceinline__ float sigm(float x) { return 1.0f / (1.0f + __expf(-x)); }

__device__ __forceinline__ float wave_sum(float v) {
#pragma unroll
    for (int o = 16; o > 0; o >>= 1) v += __shfl_xor(v, o, 32);
    return v;
}

// Async copy: 16 bytes per lane, global -> LDS, tracked with ASYNCcnt.
__device__ __forceinline__ void async_copy16(unsigned lds_off, const void* gaddr) {
    asm volatile("global_load_async_to_lds_b128 %0, %1, off"
                 :: "v"(lds_off), "v"((unsigned long long)(uintptr_t)gaddr)
                 : "memory");
}
__device__ __forceinline__ void wait_async_le4() {
    asm volatile("s_wait_asynccnt 0x4" ::: "memory");
}
__device__ __forceinline__ void wait_async_0() {
    asm volatile("s_wait_asynccnt 0x0" ::: "memory");
}

// Load a 16x32 bf16 fragment (A or B operand) from an LDS [row][k] tile.
// ISA 16-bit A-matrix layout: lane l holds row (l&15); VGPR q holds K-pair
// at offset ((q<4)?2q:16+2(q-4)) + 8*(l>=16).
__device__ __forceinline__ v16bf load_frag16(const unsigned short* base, int stride,
                                             int row, int k_base, int lane) {
    union { v16bf v; unsigned u[8]; } r;
    const unsigned short* p = base + (size_t)row * stride + k_base;
    const int hi = (lane & 16) ? 8 : 0;
#pragma unroll
    for (int q = 0; q < 8; ++q) {
        int off = ((q < 4) ? (2 * q) : (16 + 2 * (q - 4))) + hi;
        r.u[q] = *(const unsigned*)(p + off);
    }
    return r.v;
}

__device__ __forceinline__ v8f wmma_bf16(v16bf a, v16bf b, v8f c) {
    return __builtin_amdgcn_wmma_f32_16x16x32_bf16(false, a, false, b, (short)0, c, false, false);
}

// ---------------- Kernel 0: convert 6 weight matrices f32 -> bf16 ----------
__global__ void w2bf_kernel(const float* w0, const float* w1, const float* w2,
                            const float* w3, const float* w4, const float* w5,
                            unsigned short* dst) {
    const float* ws[6] = {w0, w1, w2, w3, w4, w5};
    int tid = blockIdx.x * blockDim.x + threadIdx.x;   // 6*16384 threads
    dst[tid] = f2bf(ws[tid >> 14][tid & 16383]);
}

// ---------------- Kernel 1: LayerNorm(x) -> bf16 xn ------------------------
__global__ void ln_kernel(const float* __restrict__ x, const float* __restrict__ g,
                          const float* __restrict__ b, unsigned short* __restrict__ xn) {
    int wave = threadIdx.x >> 5, lane = threadIdx.x & 31;
    size_t row = (size_t)blockIdx.x * 8 + wave;
    const float4 v = *(const float4*)(x + row * DIM + lane * 4);
    float s  = wave_sum(v.x + v.y + v.z + v.w);
    float sq = wave_sum(v.x*v.x + v.y*v.y + v.z*v.z + v.w*v.w);
    float mu = s * (1.0f / DIM);
    float var = sq * (1.0f / DIM) - mu * mu;
    float rs = rsqrtf(var + 1e-5f);
    int d = lane * 4;
    float4 gg = *(const float4*)(g + d);
    float4 bb = *(const float4*)(b + d);
    US4 o;
    o.x = f2bf((v.x - mu) * rs * gg.x + bb.x);
    o.y = f2bf((v.y - mu) * rs * gg.y + bb.y);
    o.z = f2bf((v.z - mu) * rs * gg.z + bb.z);
    o.w = f2bf((v.w - mu) * rs * gg.w + bb.w);
    *(US4*)(xn + row * DIM + d) = o;
}

// ---------------- Kernel 2: 5-way projection GEMM + gates + mask -----------
// xn (NN x 128) @ {Wl,Wr,Wlg,Wrg,Wog}; emits rightT2[d][i][k] / leftT2[d][j][k]
// (k contiguous, async-copy-friendly for the einsum) and out_gate row-major f32.
__global__ void proj_kernel(const unsigned short* __restrict__ xn,
                            const int* __restrict__ src_mask,
                            const unsigned short* __restrict__ wbf, // 5 matrices [k][n]
                            const float* __restrict__ bl, const float* __restrict__ br,
                            const float* __restrict__ blg, const float* __restrict__ brg,
                            const float* __restrict__ bog,
                            unsigned short* __restrict__ leftT2,
                            unsigned short* __restrict__ rightT2,
                            float* __restrict__ og) {
    __shared__ unsigned short lds_x[128 * 128];     // A tile [m][k]
    __shared__ unsigned short lds_w[5 * 16 * 128];  // B tiles transposed [w][n][k]
    const int tid = threadIdx.x;
    const size_t r0 = (size_t)blockIdx.x * 128;

    // stage xn tile via async global->LDS (contiguous 32KB)
    {
        unsigned ldsx = (unsigned)(uintptr_t)&lds_x[0];
        const char* src = (const char*)(xn + r0 * DIM);
#pragma unroll
        for (int c = 0; c < 8; ++c) {
            int f = tid + c * 256;                   // 2048 units of 16B
            async_copy16(ldsx + f * 16, src + (size_t)f * 16);
        }
        wait_async_0();
    }
    __syncthreads();

    const int wave = tid >> 5, lane = tid & 31;
    const int m0 = wave * 16;
    const int nlan = lane & 15, mh = (lane & 16) ? 8 : 0;
    const int p = (int)(r0 >> 9);                    // constant within block
    const float mp = (float)src_mask[p];

    for (int nt = 0; nt < 8; ++nt) {
        // stage transposed weight slabs for this n-tile
        for (int idx = tid; idx < 5 * 2048; idx += 256) {
            int w = idx >> 11, r2 = idx & 2047, k = r2 >> 4, nl = r2 & 15;
            lds_w[w * 2048 + nl * 128 + k] = wbf[w * 16384 + k * 128 + nt * 16 + nl];
        }
        __syncthreads();

        v8f aL = {}, aR = {}, aLG = {}, aRG = {}, aOG = {};
#pragma unroll
        for (int kt = 0; kt < 4; ++kt) {
            v16bf af = load_frag16(lds_x, 128, m0 + nlan, kt * 32, lane);
            v16bf b0 = load_frag16(lds_w + 0 * 2048, 128, nlan, kt * 32, lane);
            v16bf b1 = load_frag16(lds_w + 1 * 2048, 128, nlan, kt * 32, lane);
            v16bf b2 = load_frag16(lds_w + 2 * 2048, 128, nlan, kt * 32, lane);
            v16bf b3 = load_frag16(lds_w + 3 * 2048, 128, nlan, kt * 32, lane);
            v16bf b4 = load_frag16(lds_w + 4 * 2048, 128, nlan, kt * 32, lane);
            aL  = wmma_bf16(af, b0, aL);
            aR  = wmma_bf16(af, b1, aR);
            aLG = wmma_bf16(af, b2, aLG);
            aRG = wmma_bf16(af, b3, aRG);
            aOG = wmma_bf16(af, b4, aOG);
        }
        const int n = nt * 16 + nlan;                // channel d
        const float blv = bl[n], brv = br[n], blgv = blg[n], brgv = brg[n], bogv = bog[n];
#pragma unroll
        for (int vr = 0; vr < 8; ++vr) {
            int mrow = m0 + mh + vr;
            size_t r = r0 + mrow;
            int q = (int)(r & 511);
            float mk = mp * (float)src_mask[q];
            float lv = (aL[vr] + blv) * mk * sigm(aLG[vr] + blgv);
            float rv = (aR[vr] + brv) * mk * sigm(aRG[vr] + brgv);
            float gv = sigm(aOG[vr] + bogv);
            // [d][col][k] layouts: col = q, k = p
            size_t cidx = (size_t)n * NN + (size_t)q * NSEQ + p;
            leftT2[cidx]  = f2bf(lv);
            rightT2[cidx] = f2bf(rv);
            og[r * DIM + n] = gv;
        }
        __syncthreads();
    }
}

// ---------------- Kernel 3: triangle einsum, per-channel GEMM --------------
// out[i,j,d] = sum_k rightT2[d][i][k] * leftT2[d][j][k]
// Double-buffered async global->LDS staging (ASYNCcnt pipeline), WMMA bf16.
__global__ void tri_kernel(const unsigned short* __restrict__ leftT2,
                           const unsigned short* __restrict__ rightT2,
                           float* __restrict__ out) {
    __shared__ unsigned short bufA[2][64 * 32];   // [i][k] slabs
    __shared__ unsigned short bufB[2][64 * 32];   // [j][k] slabs
    const int bid = blockIdx.x;
    const int t = bid & 63;
    const int d = bid >> 6;
    const int i0 = (t >> 3) * 64, j0 = (t & 7) * 64;
    const int tid = threadIdx.x;                // 128 threads, 4 waves
    const int wave = tid >> 5, lane = tid & 31;
    const int nlan = lane & 15, mh = (lane & 16) ? 8 : 0;
    const size_t cb = (size_t)d * NN;

    const unsigned short* gA = rightT2 + cb + (size_t)i0 * NSEQ;  // + col*512 + k
    const unsigned short* gB = leftT2  + cb + (size_t)j0 * NSEQ;
    const unsigned ldsA[2] = {(unsigned)(uintptr_t)&bufA[0][0], (unsigned)(uintptr_t)&bufA[1][0]};
    const unsigned ldsB[2] = {(unsigned)(uintptr_t)&bufB[0][0], (unsigned)(uintptr_t)&bufB[1][0]};

    // thread -> two 16B units per operand slab per K-step
    const int u0 = tid, u1 = tid + 128;           // unit: col = u>>2, part = u&3
    const int c0 = u0 >> 2, p0 = u0 & 3;
    const int c1 = u1 >> 2, p1 = u1 & 3;

#define STAGE(bufsel, k0)                                                         \
    do {                                                                          \
        async_copy16(ldsA[bufsel] + (u0 * 16), gA + (size_t)c0 * NSEQ + (k0) + p0 * 8); \
        async_copy16(ldsB[bufsel] + (u0 * 16), gB + (size_t)c0 * NSEQ + (k0) + p0 * 8); \
        async_copy16(ldsA[bufsel] + (u1 * 16), gA + (size_t)c1 * NSEQ + (k0) + p1 * 8); \
        async_copy16(ldsB[bufsel] + (u1 * 16), gB + (size_t)c1 * NSEQ + (k0) + p1 * 8); \
    } while (0)

    v8f acc0 = {}, acc1 = {}, acc2 = {}, acc3 = {};
    STAGE(0, 0);
    for (int kt = 0; kt < 16; ++kt) {
        const int cur = kt & 1;
        if (kt + 1 < 16) {
            STAGE(1 - cur, (kt + 1) * 32);
            wait_async_le4();                      // current slab resident (in-order)
        } else {
            wait_async_0();
        }
        __syncthreads();                           // all waves' fills visible
        const unsigned short* A = bufA[cur];
        const unsigned short* B = bufB[cur];
        v16bf af = load_frag16(A, 32, wave * 16 + nlan, 0, lane);
        v16bf b0 = load_frag16(B, 32,  0 + nlan, 0, lane);
        v16bf b1 = load_frag16(B, 32, 16 + nlan, 0, lane);
        v16bf b2 = load_frag16(B, 32, 32 + nlan, 0, lane);
        v16bf b3 = load_frag16(B, 32, 48 + nlan, 0, lane);
        acc0 = wmma_bf16(af, b0, acc0);
        acc1 = wmma_bf16(af, b1, acc1);
        acc2 = wmma_bf16(af, b2, acc2);
        acc3 = wmma_bf16(af, b3, acc3);
        __syncthreads();                           // protect buf[cur] from next prefetch
    }
#undef STAGE

    const int irow = i0 + wave * 16 + mh;
#pragma unroll
    for (int jt = 0; jt < 4; ++jt) {
        v8f a = (jt == 0) ? acc0 : (jt == 1) ? acc1 : (jt == 2) ? acc2 : acc3;
        int j = j0 + jt * 16 + nlan;
#pragma unroll
        for (int vr = 0; vr < 8; ++vr)
            out[((size_t)(irow + vr) * NSEQ + j) * DIM + d] = a[vr];
    }
}

// ---------------- Kernel 4: out-LN * gate, then @ Wo + bo (in-place) -------
__global__ void out_kernel(float* __restrict__ io,          // d_out: raw in, final out
                           const float* __restrict__ og,
                           const unsigned short* __restrict__ wo_bf,
                           const float* __restrict__ oln_g, const float* __restrict__ oln_b,
                           const float* __restrict__ bo) {
    __shared__ unsigned short lds_x[128 * 128];
    __shared__ unsigned short lds_w[16 * 128];
    const int tid = threadIdx.x;
    const size_t r0 = (size_t)blockIdx.x * 128;
    const int wave = tid >> 5, lane = tid & 31;
    const int nlan = lane & 15, mh = (lane & 16) ? 8 : 0;

    // phase 1: LN + gate -> bf16 LDS tile (each wave: 16 rows)
    const int dd = lane * 4;
    const float4 gg = *(const float4*)(oln_g + dd);
    const float4 bb = *(const float4*)(oln_b + dd);
#pragma unroll
    for (int j = 0; j < 16; ++j) {
        int ml = wave * 16 + j;
        size_t r = r0 + ml;
        float4 v = *(const float4*)(io + r * DIM + dd);
        float s  = wave_sum(v.x + v.y + v.z + v.w);
        float sq = wave_sum(v.x*v.x + v.y*v.y + v.z*v.z + v.w*v.w);
        float mu = s * (1.0f / DIM);
        float rs = rsqrtf(sq * (1.0f / DIM) - mu * mu + 1e-5f);
        float4 gt = *(const float4*)(og + r * DIM + dd);
        lds_x[ml * 128 + dd + 0] = f2bf(((v.x - mu) * rs * gg.x + bb.x) * gt.x);
        lds_x[ml * 128 + dd + 1] = f2bf(((v.y - mu) * rs * gg.y + bb.y) * gt.y);
        lds_x[ml * 128 + dd + 2] = f2bf(((v.z - mu) * rs * gg.z + bb.z) * gt.z);
        lds_x[ml * 128 + dd + 3] = f2bf(((v.w - mu) * rs * gg.w + bb.w) * gt.w);
    }
    __syncthreads();

    // phase 2: GEMM with Wo
    const int m0 = wave * 16;
    for (int nt = 0; nt < 8; ++nt) {
        for (int idx = tid; idx < 2048; idx += 256) {
            int k = idx >> 4, nl = idx & 15;
            lds_w[nl * 128 + k] = wo_bf[k * 128 + nt * 16 + nl];
        }
        __syncthreads();
        v8f acc = {};
#pragma unroll
        for (int kt = 0; kt < 4; ++kt) {
            v16bf af = load_frag16(lds_x, 128, m0 + nlan, kt * 32, lane);
            v16bf bf = load_frag16(lds_w, 128, nlan, kt * 32, lane);
            acc = wmma_bf16(af, bf, acc);
        }
        int n = nt * 16 + nlan;
        float bov = bo[n];
#pragma unroll
        for (int vr = 0; vr < 8; ++vr) {
            size_t r = r0 + m0 + mh + vr;
            io[r * DIM + n] = acc[vr] + bov;
        }
        __syncthreads();
    }
}

extern "C" void kernel_launch(void* const* d_in, const int* in_sizes, int n_in,
                              void* d_out, int out_size, void* d_ws, size_t ws_size,
                              hipStream_t stream) {
    (void)in_sizes; (void)n_in; (void)out_size; (void)ws_size;
    const float* x        = (const float*)d_in[0];
    const int*   src_mask = (const int*)  d_in[1];
    const float* ln_g = (const float*)d_in[2];
    const float* ln_b = (const float*)d_in[3];
    const float* Wl  = (const float*)d_in[4];   const float* bl  = (const float*)d_in[5];
    const float* Wr  = (const float*)d_in[6];   const float* br  = (const float*)d_in[7];
    const float* Wlg = (const float*)d_in[8];   const float* blg = (const float*)d_in[9];
    const float* Wrg = (const float*)d_in[10];  const float* brg = (const float*)d_in[11];
    const float* Wog = (const float*)d_in[12];  const float* bog = (const float*)d_in[13];
    const float* oln_g = (const float*)d_in[14];
    const float* oln_b = (const float*)d_in[15];
    const float* Wo  = (const float*)d_in[16];  const float* bo  = (const float*)d_in[17];
    float* out = (float*)d_out;

    // workspace layout
    char* ws = (char*)d_ws;
    unsigned short* xn      = (unsigned short*)ws;                ws += (size_t)NN * DIM * 2;
    unsigned short* wbf     = (unsigned short*)ws;                ws += (size_t)6 * DIM * DIM * 2;
    unsigned short* leftT2  = (unsigned short*)ws;                ws += (size_t)NN * DIM * 2;
    unsigned short* rightT2 = (unsigned short*)ws;                ws += (size_t)NN * DIM * 2;
    float*          og      = (float*)ws;                         ws += (size_t)NN * DIM * 4;

    w2bf_kernel<<<6 * DIM * DIM / 256, 256, 0, stream>>>(Wl, Wr, Wlg, Wrg, Wog, Wo, wbf);
    ln_kernel<<<NN / 8, 256, 0, stream>>>(x, ln_g, ln_b, xn);
    proj_kernel<<<NN / 128, 256, 0, stream>>>(xn, src_mask, wbf, bl, br, blg, brg, bog,
                                              leftT2, rightT2, og);
    tri_kernel<<<64 * DIM, 128, 0, stream>>>(leftT2, rightT2, out);
    out_kernel<<<NN / 128, 256, 0, stream>>>(out, og, wbf + 5 * DIM * DIM, oln_g, oln_b, bo);
}
